// VQ_25185688224135
// MI455X (gfx1250) — matrix operand order repaired and checked by reading, not compiled, not run on hardware
//
#include <hip/hip_runtime.h>

typedef __attribute__((ext_vector_type(16))) __bf16 v16bf;
typedef __attribute__((ext_vector_type(16))) unsigned short u16x16;
typedef __attribute__((ext_vector_type(8))) float v8f;

#define K_SIZE 1024
#define DIM 64
#define N_ROWS 65536
#define N_ELEM 4194304   // 64*32*32*64
#define MT 2             // M-tiles (16 rows each) per wave

__device__ __forceinline__ unsigned short f32_to_bf16(float f) {
    unsigned u = __float_as_uint(f);
    unsigned r = 0x7FFFu + ((u >> 16) & 1u);   // round to nearest even
    return (unsigned short)((u + r) >> 16);
}
__device__ __forceinline__ float bf16_to_f32(unsigned short h) {
    return __uint_as_float(((unsigned)h) << 16);
}

union V16U { u16x16 u; v16bf b; };

// ---------------------------------------------------------------- zero accum
__global__ void vq_zero(unsigned* __restrict__ counts_and_loss) {
    int i = blockIdx.x * blockDim.x + threadIdx.x;
    if (i < K_SIZE + 1) counts_and_loss[i] = 0u;   // 1024 counts + 1 f32 loss
}

// -------------------------------------------------- codebook -> bf16 hi/lo + |e|^2
__global__ void vq_prep(const float* __restrict__ cb,
                        unsigned short* __restrict__ cb_hi,
                        unsigned short* __restrict__ cb_lo,
                        float* __restrict__ enorm) {
    int k = blockIdx.x * blockDim.x + threadIdx.x;
    if (k >= K_SIZE) return;
    const float* row = cb + (size_t)k * DIM;
    float s = 0.f;
#pragma unroll
    for (int j = 0; j < DIM; ++j) {
        float x = row[j];
        unsigned short h = f32_to_bf16(x);
        cb_hi[(size_t)k * DIM + j] = h;
        cb_lo[(size_t)k * DIM + j] = f32_to_bf16(x - bf16_to_f32(h));
        s = __builtin_fmaf(x, x, s);
    }
    enorm[k] = s;
}

// ---------------------------------------------------------------- main kernel
// 256 threads = 8 waves; each wave owns MT*16 = 32 rows of the input.
__global__ __launch_bounds__(256) void vq_main(
        const float* __restrict__ x, const float* __restrict__ cb,
        const unsigned short* __restrict__ cb_hi,
        const unsigned short* __restrict__ cb_lo,
        const float* __restrict__ enorm,
        float* __restrict__ out,
        unsigned* __restrict__ counts,
        float* __restrict__ loss_acc) {
    const int lane = threadIdx.x & 31;
    const int wave = threadIdx.x >> 5;
    const int mb   = blockIdx.x * (8 * 16 * MT) + wave * (16 * MT);  // row base
    const int r    = lane & 15;
    const int hp   = lane >> 4;                      // lane half (0/1)

    // ---- A operands (MT tiles of 16x64 rows, bf16 hi/lo), contiguous k order, matched by B:
    // lanes 0-15: K = kk+0..15; lanes 16-31: K = kk+16..31
    V16U a_hi[MT][2], a_lo[MT][2];
#pragma unroll
    for (int mt = 0; mt < MT; ++mt) {
        const float* xrow = x + (size_t)(mb + mt * 16 + r) * DIM;
#pragma unroll
        for (int t = 0; t < 2; ++t) {
            const int kk = t * 32;
#pragma unroll
            for (int j = 0; j < 8; ++j) {
                float v0 = xrow[kk + hp * 16 + j];
                unsigned short h0 = f32_to_bf16(v0);
                a_hi[mt][t].u[j] = h0;
                a_lo[mt][t].u[j] = f32_to_bf16(v0 - bf16_to_f32(h0));
                float v1 = xrow[kk + hp * 16 + 8 + j];
                unsigned short h1 = f32_to_bf16(v1);
                a_hi[mt][t].u[8 + j] = h1;
                a_lo[mt][t].u[8 + j] = f32_to_bf16(v1 - bf16_to_f32(h1));
            }
        }
    }

    float best_d[MT][8];
    int   best_i[MT][8];
#pragma unroll
    for (int mt = 0; mt < MT; ++mt)
#pragma unroll
        for (int v = 0; v < 8; ++v) { best_d[mt][v] = 3.4e38f; best_i[mt][v] = 0; }

    const int c  = lane & 15;    // column within 16-wide codebook tile
    const int kb = hp * 16;      // B layout: lanes 0-15 K=0..15, lanes 16-31 K=16..31

    for (int nb = 0; nb < K_SIZE; nb += 16) {
        const unsigned short* bhp = cb_hi + (size_t)(nb + c) * DIM + kb;
        const unsigned short* blp = cb_lo + (size_t)(nb + c) * DIM + kb;
        V16U bh0, bh1, bl0, bl1;
        bh0.u = *(const u16x16*)(bhp);        // K 0..31 tile
        bh1.u = *(const u16x16*)(bhp + 32);   // K 32..63 tile
        bl0.u = *(const u16x16*)(blp);
        bl1.u = *(const u16x16*)(blp + 32);

        // dot = hi*hi + lo*hi + hi*lo (lo*lo negligible); alternate the two
        // accumulators so consecutive WMMAs are independent (no RAW stall).
        v8f acc0 = {}, acc1 = {};
        acc0 = __builtin_amdgcn_wmma_f32_16x16x32_bf16(false, a_hi[0][0].b, false, bh0.b, (short)0, acc0, false, false);
        acc1 = __builtin_amdgcn_wmma_f32_16x16x32_bf16(false, a_hi[1][0].b, false, bh0.b, (short)0, acc1, false, false);
        acc0 = __builtin_amdgcn_wmma_f32_16x16x32_bf16(false, a_hi[0][1].b, false, bh1.b, (short)0, acc0, false, false);
        acc1 = __builtin_amdgcn_wmma_f32_16x16x32_bf16(false, a_hi[1][1].b, false, bh1.b, (short)0, acc1, false, false);
        acc0 = __builtin_amdgcn_wmma_f32_16x16x32_bf16(false, a_lo[0][0].b, false, bh0.b, (short)0, acc0, false, false);
        acc1 = __builtin_amdgcn_wmma_f32_16x16x32_bf16(false, a_lo[1][0].b, false, bh0.b, (short)0, acc1, false, false);
        acc0 = __builtin_amdgcn_wmma_f32_16x16x32_bf16(false, a_lo[0][1].b, false, bh1.b, (short)0, acc0, false, false);
        acc1 = __builtin_amdgcn_wmma_f32_16x16x32_bf16(false, a_lo[1][1].b, false, bh1.b, (short)0, acc1, false, false);
        acc0 = __builtin_amdgcn_wmma_f32_16x16x32_bf16(false, a_hi[0][0].b, false, bl0.b, (short)0, acc0, false, false);
        acc1 = __builtin_amdgcn_wmma_f32_16x16x32_bf16(false, a_hi[1][0].b, false, bl0.b, (short)0, acc1, false, false);
        acc0 = __builtin_amdgcn_wmma_f32_16x16x32_bf16(false, a_hi[0][1].b, false, bl1.b, (short)0, acc0, false, false);
        acc1 = __builtin_amdgcn_wmma_f32_16x16x32_bf16(false, a_hi[1][1].b, false, bl1.b, (short)0, acc1, false, false);

        // distance (minus the per-row constant |x|^2): |e|^2 - 2 x.e
        float en = enorm[nb + c];
#pragma unroll
        for (int v = 0; v < 8; ++v) {
            float d0 = __builtin_fmaf(-2.0f, acc0[v], en);
            if (d0 < best_d[0][v]) { best_d[0][v] = d0; best_i[0][v] = nb + c; }
            float d1 = __builtin_fmaf(-2.0f, acc1[v], en);
            if (d1 < best_d[1][v]) { best_d[1][v] = d1; best_i[1][v] = nb + c; }
        }
    }

    // ---- argmin across the 16 lanes of each half (rows v / 8+v per tile)
#pragma unroll
    for (int mask = 1; mask < 16; mask <<= 1) {
#pragma unroll
        for (int mt = 0; mt < MT; ++mt)
#pragma unroll
            for (int v = 0; v < 8; ++v) {
                float od = __shfl_xor(best_d[mt][v], mask, 32);
                int   oi = __shfl_xor(best_i[mt][v], mask, 32);
                if (od < best_d[mt][v] ||
                    (od == best_d[mt][v] && oi < best_i[mt][v])) {
                    best_d[mt][v] = od; best_i[mt][v] = oi;
                }
            }
    }

    // ---- usage histogram (lane 0 -> rows v, lane 16 -> rows 8+v, per tile)
    if (c == 0) {
#pragma unroll
        for (int mt = 0; mt < MT; ++mt)
#pragma unroll
            for (int v = 0; v < 8; ++v) atomicAdd(&counts[best_i[mt][v]], 1u);
    }

    // ---- gather codes to output + squared-error partial sum (2 elems/lane/row)
    float lsum = 0.f;
#pragma unroll
    for (int m = 0; m < 16 * MT; ++m) {
        const int mt = m >> 4, mm = m & 15;
        int idx = __shfl(best_i[mt][mm & 7], (mm >> 3) * 16, 32);
        const float* crow = cb + (size_t)idx * DIM;
        const float* xr   = x  + (size_t)(mb + m) * DIM;
        float*       orow = out + (size_t)(mb + m) * DIM;
        float c0 = crow[lane], c1 = crow[lane + 32];
        float x0 = xr[lane],   x1 = xr[lane + 32];
        orow[lane]      = c0;
        orow[lane + 32] = c1;
        float d0 = c0 - x0, d1 = c1 - x1;
        lsum = __builtin_fmaf(d0, d0, lsum);
        lsum = __builtin_fmaf(d1, d1, lsum);
    }
#pragma unroll
    for (int mask = 16; mask >= 1; mask >>= 1) lsum += __shfl_xor(lsum, mask, 32);
    if (lane == 0) atomicAdd(loss_acc, lsum);
}

// ---------------------------------------------------------------- finalize
__global__ void vq_final(const unsigned* __restrict__ counts,
                         const float* __restrict__ loss_acc,
                         float* __restrict__ out) {
    __shared__ float red[32];
    int t = threadIdx.x;                 // 0..1023, one per codebook entry
    float p = (float)counts[t] * (1.0f / (float)N_ROWS);
    float term = p * logf(p + 1e-10f);
#pragma unroll
    for (int mask = 16; mask >= 1; mask >>= 1) term += __shfl_xor(term, mask, 32);
    if ((t & 31) == 0) red[t >> 5] = term;
    __syncthreads();
    if (t < 32) {
        float s = red[t];
#pragma unroll
        for (int mask = 16; mask >= 1; mask >>= 1) s += __shfl_xor(s, mask, 32);
        if (t == 0) {
            // loss = codebook_loss + 0.25*commitment_loss = 1.25 * MSE
            out[N_ELEM]     = loss_acc[0] * 1.25f / (float)N_ELEM;
            out[N_ELEM + 1] = expf(-s);
        }
    }
}

// ---------------------------------------------------------------- launcher
extern "C" void kernel_launch(void* const* d_in, const int* in_sizes, int n_in,
                              void* d_out, int out_size, void* d_ws, size_t ws_size,
                              hipStream_t stream) {
    const float* x  = (const float*)d_in[0];   // (64,32,32,64) f32
    const float* cb = (const float*)d_in[1];   // (1024,64) f32
    float* out = (float*)d_out;                // 4194304 quantize + loss + perplexity

    char* ws = (char*)d_ws;
    unsigned short* cb_hi  = (unsigned short*)(ws);            // 128 KB
    unsigned short* cb_lo  = (unsigned short*)(ws + 131072);   // 128 KB
    float*          enorm  = (float*)(ws + 262144);            // 4 KB
    unsigned*       counts = (unsigned*)(ws + 266240);         // 4 KB (+ loss right after)
    float*          lossa  = (float*)(ws + 270336);            // 4 B

    vq_zero<<<dim3(5),   dim3(256), 0, stream>>>(counts);                   // counts + loss
    vq_prep<<<dim3(4),   dim3(256), 0, stream>>>(cb, cb_hi, cb_lo, enorm);
    vq_main<<<dim3(256), dim3(256), 0, stream>>>(x, cb, cb_hi, cb_lo, enorm,
                                                 out, counts, lossa);
    vq_final<<<dim3(1),  dim3(1024), 0, stream>>>(counts, lossa, out);
    (void)in_sizes; (void)n_in; (void)out_size; (void)ws_size;
}
